// EfficientUpdateFormer_53884659695692
// MI455X (gfx1250) — compile-verified, matrix operand order checked by reading, not run on hardware
//
#include <hip/hip_runtime.h>
#include <hip/hip_bf16.h>

typedef unsigned short u16;
typedef __attribute__((ext_vector_type(16))) __bf16 v16bf;
typedef __attribute__((ext_vector_type(8)))  float  v8f;
typedef __attribute__((ext_vector_type(4)))  unsigned int uintx4;
typedef int gvec4i __attribute__((vector_size(16)));   // matches builtin param type

union FragBF {
    v16bf v;
    uintx4 u[2];
    u16 s[16];
};

#define NEGBIG (-1.0e9f)

// Probe for the gfx1250 async global->LDS path (ASYNCcnt-tracked).
#if defined(__has_builtin)
#  if __has_builtin(__builtin_amdgcn_global_load_async_to_lds_b128) && \
      __has_builtin(__builtin_amdgcn_s_wait_asynccnt)
#    define HAVE_ASYNC_LDS 1
#  endif
#endif
#ifndef HAVE_ASYNC_LDS
#  define HAVE_ASYNC_LDS 0
#endif

__device__ __forceinline__ u16 f2bf(float f) {
    unsigned u = __float_as_uint(f);
    unsigned r = u + 0x7FFFu + ((u >> 16) & 1u);
    return (u16)(r >> 16);
}

__device__ __forceinline__ float gelu_f(float x) {
    float x3 = x * x * x;
    return 0.5f * x * (1.0f + tanhf(0.7978845608f * (x + 0.044715f * x3)));
}

// ---------------------------------------------------------------------------
// Tiled bf16 WMMA GEMM:  C[M,N] = act(A[M,K] * B^T + bias) + res
//   A: bf16 row-major M x K (lda)
//   B: bf16 row-major N x K (ldb)  (i.e. pre-transposed weights)
// Epilogue: *alpha, +bias[n], gelu, +res[m,n], store f32 and/or bf16.
// ---------------------------------------------------------------------------
struct GemmArgs {
    const u16* A;
    const u16* B;
    const float* bias;
    const float* res;
    float* Cf;
    u16*   Cb;
    long lda, ldb, ldr, ldc, ldcb;
    int M, N, K, gelu;
    float alpha;
};

#define BKP 40  // 32 + 8 pad (keeps rows 16B aligned, spreads banks)

__device__ __forceinline__ void stage_mat(const u16* src, long ld, int bound,
                                          int r0, int k0, int K,
                                          u16 (*dst)[BKP], int tid) {
    for (int idx = tid; idx < 512; idx += 256) {
        int r  = idx >> 2;
        int c8 = (idx & 3) << 3;
        int gr = r0 + r, gk = k0 + c8;
        union { uintx4 v; u16 s[8]; } t;
        if (gr < bound && (gk + 8) <= K) {
            t.v = *(const uintx4*)(src + (long)gr * ld + gk);
        } else {
            for (int e = 0; e < 8; ++e)
                t.s[e] = (gr < bound && (gk + e) < K) ? src[(long)gr * ld + gk + e] : (u16)0;
        }
        *(uintx4*)&dst[r][c8] = t.v;
    }
}

// Stage one 128x32 bf16 tile; interior tiles use async global->LDS DMA when
// the toolchain exposes it (tracked by ASYNCcnt), edges fall back to VGPR path.
__device__ __forceinline__ void stage_any(const u16* src, long ld, int bound,
                                          int r0, int k0, int K,
                                          u16 (*dst)[BKP], int tid) {
#if HAVE_ASYNC_LDS
    if ((r0 + 128) <= bound && (k0 + 32) <= K) {
        for (int idx = tid; idx < 512; idx += 256) {
            int r  = idx >> 2;
            int c8 = (idx & 3) << 3;
            const u16* g = src + (long)(r0 + r) * ld + (k0 + c8);
            __attribute__((address_space(1))) gvec4i* gp =
                (__attribute__((address_space(1))) gvec4i*)
                (__attribute__((address_space(1))) const void*)g;
            __attribute__((address_space(3))) gvec4i* lp =
                (__attribute__((address_space(3))) gvec4i*)
                (__attribute__((address_space(3))) void*)&dst[r][c8];
            __builtin_amdgcn_global_load_async_to_lds_b128(gp, lp, 0, 0);
        }
        return;
    }
#endif
    stage_mat(src, ld, bound, r0, k0, K, dst, tid);
}

__device__ __forceinline__ void wait_async_stage() {
#if HAVE_ASYNC_LDS
    __builtin_amdgcn_s_wait_asynccnt(0);
#endif
}

__global__ __launch_bounds__(256) void k_gemm(GemmArgs g) {
    __shared__ u16 lA[2][128][BKP];
    __shared__ u16 lB[2][128][BKP];

    int tid = threadIdx.x;
    int m0 = blockIdx.y * 128, n0 = blockIdx.x * 128;
    int wave = tid >> 5, lane = tid & 31;
    int wm = wave & 3, wn = wave >> 2;          // 4 M-waves x 2 N-waves
    int hi = lane >> 4, l16 = lane & 15, kh = hi * 8;

    v8f acc[2][4];
    for (int i = 0; i < 2; ++i)
        for (int j = 0; j < 4; ++j)
            for (int e = 0; e < 8; ++e) acc[i][j][e] = 0.0f;

    int nkt = (g.K + 31) >> 5;

    stage_any(g.A, g.lda, g.M, m0, 0, g.K, lA[0], tid);
    stage_any(g.B, g.ldb, g.N, n0, 0, g.K, lB[0], tid);

    for (int t = 0; t < nkt; ++t) {
        wait_async_stage();
        __syncthreads();
        if (t + 1 < nkt) {
            if (t + 2 < nkt) {
                __builtin_prefetch(g.A + (long)m0 * g.lda + (long)(t + 2) * 32, 0, 1);
                __builtin_prefetch(g.B + (long)n0 * g.ldb + (long)(t + 2) * 32, 0, 1);
            }
            stage_any(g.A, g.lda, g.M, m0, (t + 1) * 32, g.K, lA[(t + 1) & 1], tid);
            stage_any(g.B, g.ldb, g.N, n0, (t + 1) * 32, g.K, lB[(t + 1) & 1], tid);
        }
        int cur = t & 1;
        FragBF af[2], bfr[4];
        for (int fm = 0; fm < 2; ++fm) {
            int m = wm * 32 + fm * 16 + l16;
            af[fm].u[0] = *(const uintx4*)&lA[cur][m][kh];
            af[fm].u[1] = *(const uintx4*)&lA[cur][m][16 + kh];
        }
        for (int fn = 0; fn < 4; ++fn) {
            int n = wn * 64 + fn * 16 + l16;
            bfr[fn].u[0] = *(const uintx4*)&lB[cur][n][kh];
            bfr[fn].u[1] = *(const uintx4*)&lB[cur][n][16 + kh];
        }
        for (int fm = 0; fm < 2; ++fm)
            for (int fn = 0; fn < 4; ++fn)
                acc[fm][fn] = __builtin_amdgcn_wmma_f32_16x16x32_bf16(
                    false, af[fm].v, false, bfr[fn].v, (short)0, acc[fm][fn], false, false);
    }

    int wrow = m0 + wm * 32, wcol = n0 + wn * 64;
    for (int fm = 0; fm < 2; ++fm) {
        for (int fn = 0; fn < 4; ++fn) {
            int nn = wcol + fn * 16 + l16;
            if (nn >= g.N) continue;
            float bv = g.bias ? g.bias[nn] : 0.0f;
            for (int e = 0; e < 8; ++e) {
                int mm = wrow + fm * 16 + e + hi * 8;
                if (mm >= g.M) continue;
                float v = acc[fm][fn][e] * g.alpha + bv;
                if (g.gelu) v = gelu_f(v);
                if (g.res) v += g.res[(long)mm * g.ldr + nn];
                if (g.Cf) g.Cf[(long)mm * g.ldc + nn] = v;
                if (g.Cb) g.Cb[(long)mm * g.ldcb + nn] = f2bf(v);
            }
        }
    }
}

// ---------------------------------------------------------------------------
// Fused online-softmax attention, one wave per (seq, head, 16-query tile).
// hd <= 48, K padded to 64 for S = Q*K^T (two bf16 WMMA k-steps),
// P@V via 3 N-groups of 16 with K padded 16->32.
// ---------------------------------------------------------------------------
struct AttnArgs {
    const u16* Q;
    const u16* KV;
    u16* O;
    const float* kb;       // key-mask bias [nseq][Nk] or null
    long q_seq, kv_seq, o_seq, kb_seq;
    int q_row, kv_row, o_row;
    int k_off, v_off;
    int Nq, Nk, hd;
    float scale;
};

// stage a 16 x 64 bf16 tile (zero-padded) from global rows, 16B chunks
__device__ __forceinline__ void attn_stage64(const u16* base, long row_stride,
                                             int row0, int nrows, int hd,
                                             u16 (*dst)[72], int lane) {
    for (int i = lane; i < 16 * 8; i += 32) {
        int r = i >> 3, c8 = (i & 7) << 3;
        union { uintx4 v; u16 s[8]; } t;
        if ((row0 + r) < nrows && (c8 + 8) <= hd) {
            t.v = *(const uintx4*)(base + (long)(row0 + r) * row_stride + c8);
        } else {
            for (int e = 0; e < 8; ++e)
                t.s[e] = ((row0 + r) < nrows && (c8 + e) < hd)
                             ? base[(long)(row0 + r) * row_stride + c8 + e] : (u16)0;
        }
        *(uintx4*)&dst[r][c8] = t.v;
    }
}

__global__ __launch_bounds__(32) void k_attn(AttnArgs a) {
    __shared__ u16 lQ[16][72];
    __shared__ u16 lK[16][72];
    __shared__ u16 lV[16][56];
    __shared__ u16 lP[16][16];

    int seq = blockIdx.z, h = blockIdx.y, q0 = blockIdx.x * 16;
    int lane = threadIdx.x & 31;
    int hi = lane >> 4, l16 = lane & 15, kh = hi * 8;

    const u16* Qb = a.Q  + (long)seq * a.q_seq  + h * a.hd;
    const u16* Kb = a.KV + (long)seq * a.kv_seq + a.k_off + h * a.hd;
    const u16* Vb = a.KV + (long)seq * a.kv_seq + a.v_off + h * a.hd;

    attn_stage64(Qb, a.q_row, q0, a.Nq, a.hd, lQ, lane);
    __syncthreads();

    FragBF aq0, aq1;
    aq0.u[0] = *(const uintx4*)&lQ[l16][kh];
    aq0.u[1] = *(const uintx4*)&lQ[l16][16 + kh];
    aq1.u[0] = *(const uintx4*)&lQ[l16][32 + kh];
    aq1.u[1] = *(const uintx4*)&lQ[l16][48 + kh];

    float rmax[8], rsum[8], fcorr[8];
    v8f oacc[3];
    for (int g = 0; g < 8; ++g) {
        rmax[g] = -3.0e38f; rsum[g] = 0.0f;
        oacc[0][g] = 0.0f; oacc[1][g] = 0.0f; oacc[2][g] = 0.0f;
    }
    v8f zc;
    for (int e = 0; e < 8; ++e) zc[e] = 0.0f;

    int nkt = (a.Nk + 15) >> 4;
    for (int kt = 0; kt < nkt; ++kt) {
        int kbase = kt << 4;
        __syncthreads();
        attn_stage64(Kb, a.kv_row, kbase, a.Nk, a.hd, lK, lane);
        // V tile 16 x 48, 16B chunks (hd<=48 so all chunks are col-interior)
        for (int i = lane; i < 16 * 6; i += 32) {
            int r = i / 6, c8 = (i % 6) << 3;
            union { uintx4 v; u16 s[8]; } t;
            if ((kbase + r) < a.Nk && (c8 + 8) <= a.hd) {
                t.v = *(const uintx4*)(Vb + (long)(kbase + r) * a.kv_row + c8);
            } else {
                for (int e = 0; e < 8; ++e)
                    t.s[e] = ((kbase + r) < a.Nk && (c8 + e) < a.hd)
                                 ? Vb[(long)(kbase + r) * a.kv_row + c8 + e] : (u16)0;
            }
            *(uintx4*)&lV[r][c8] = t.v;
        }
        __syncthreads();

        FragBF bk0, bk1;
        bk0.u[0] = *(const uintx4*)&lK[l16][kh];
        bk0.u[1] = *(const uintx4*)&lK[l16][16 + kh];
        bk1.u[0] = *(const uintx4*)&lK[l16][32 + kh];
        bk1.u[1] = *(const uintx4*)&lK[l16][48 + kh];

        v8f s = __builtin_amdgcn_wmma_f32_16x16x32_bf16(false, aq0.v, false, bk0.v, (short)0, zc, false, false);
        s     = __builtin_amdgcn_wmma_f32_16x16x32_bf16(false, aq1.v, false, bk1.v, (short)0, s,  false, false);

        float bias_n = NEGBIG;
        if ((kbase + l16) < a.Nk)
            bias_n = a.kb ? a.kb[(long)seq * a.kb_seq + kbase + l16] : 0.0f;

        for (int g = 0; g < 8; ++g) {
            float v = s[g] * a.scale + bias_n;
            float mx = v;
            for (int off = 1; off < 16; off <<= 1) mx = fmaxf(mx, __shfl_xor(mx, off, 16));
            float nm = fmaxf(rmax[g], mx);
            float corr = __expf(rmax[g] - nm);
            float p = __expf(v - nm);
            float ps = p;
            for (int off = 1; off < 16; off <<= 1) ps += __shfl_xor(ps, off, 16);
            rsum[g] = rsum[g] * corr + ps;
            rmax[g] = nm;
            fcorr[g] = corr;
            lP[g + hi * 8][l16] = f2bf(p);
        }
        for (int j = 0; j < 3; ++j)
            for (int g = 0; g < 8; ++g) oacc[j][g] *= fcorr[g];
        __syncthreads();

        FragBF ap;
        ap.u[0] = *(const uintx4*)&lP[l16][kh];
        ap.u[1][0] = 0; ap.u[1][1] = 0; ap.u[1][2] = 0; ap.u[1][3] = 0;

        for (int j = 0; j < 3; ++j) {
            int col = j * 16 + l16;
            FragBF bv;
            for (int e = 0; e < 8; ++e) bv.s[e] = (col < a.hd) ? lV[kh + e][col] : (u16)0;
            for (int e = 8; e < 16; ++e) bv.s[e] = 0;
            oacc[j] = __builtin_amdgcn_wmma_f32_16x16x32_bf16(
                false, ap.v, false, bv.v, (short)0, oacc[j], false, false);
        }
    }

    u16* Ob = a.O + (long)seq * a.o_seq + h * a.hd;
    for (int j = 0; j < 3; ++j) {
        int col = j * 16 + l16;
        if (col >= a.hd) continue;
        for (int g = 0; g < 8; ++g) {
            int m = q0 + g + hi * 8;
            if (m < a.Nq) {
                float d = rsum[g];
                float val = (d > 0.0f) ? oacc[j][g] / d : 0.0f;
                Ob[(long)m * a.o_row + col] = f2bf(val);
            }
        }
    }
}

// ---------------------------------------------------------------------------
// LayerNorm (per row) -> bf16
// ---------------------------------------------------------------------------
__global__ __launch_bounds__(128) void k_layernorm(const float* x, u16* y, int C) {
    long row = blockIdx.x;
    const float* xr = x + row * (long)C;
    u16* yr = y + row * (long)C;
    __shared__ float sbuf[128];

    float s = 0.0f, ss = 0.0f;
    for (int i = threadIdx.x; i < C; i += 128) {
        float v = xr[i];
        s += v; ss += v * v;
    }
    sbuf[threadIdx.x] = s; __syncthreads();
    for (int o = 64; o > 0; o >>= 1) {
        if (threadIdx.x < o) sbuf[threadIdx.x] += sbuf[threadIdx.x + o];
        __syncthreads();
    }
    float mean = sbuf[0] / (float)C;
    __syncthreads();
    sbuf[threadIdx.x] = ss; __syncthreads();
    for (int o = 64; o > 0; o >>= 1) {
        if (threadIdx.x < o) sbuf[threadIdx.x] += sbuf[threadIdx.x + o];
        __syncthreads();
    }
    float var = sbuf[0] / (float)C - mean * mean;
    float r = rsqrtf(var + 1e-6f);
    for (int i = threadIdx.x; i < C; i += 128)
        yr[i] = f2bf((xr[i] - mean) * r);
}

// ---------------------------------------------------------------------------
// Utility kernels
// ---------------------------------------------------------------------------
__global__ void k_cvt_bf16(const float* src, u16* dst, long n) {
    for (long i = (long)blockIdx.x * 256 + threadIdx.x; i < n; i += (long)gridDim.x * 256)
        dst[i] = f2bf(src[i]);
}

__global__ void k_mask_bias(const unsigned char* m, float* kb, long n) {
    for (long i = (long)blockIdx.x * 256 + threadIdx.x; i < n; i += (long)gridDim.x * 256)
        kb[i] = m[i] ? 0.0f : NEGBIG;
}

__global__ __launch_bounds__(256) void k_transp_bf16(const float* src, u16* dst, int R, int C) {
    __shared__ float t[32][33];
    int c0 = blockIdx.x * 32, r0 = blockIdx.y * 32;
    int tx = threadIdx.x & 31, ty = threadIdx.x >> 5;
    for (int p = 0; p < 4; ++p) {
        int r = r0 + ty + p * 8, c = c0 + tx;
        t[ty + p * 8][tx] = (r < R && c < C) ? src[(long)r * C + c] : 0.0f;
    }
    __syncthreads();
    for (int p = 0; p < 4; ++p) {
        int c = c0 + ty + p * 8, r = r0 + tx;
        if (c < C && r < R) dst[(long)c * R + r] = f2bf(t[tx][ty + p * 8]);
    }
}

// layout permute between [T][Nt][C] (bt) and [Nt][T][C] (nt)
__global__ void k_permute(const float* src, float* dst, int T, int Nt, int C, int to_nt) {
    long total = (long)T * Nt * C;
    for (long idx = (long)blockIdx.x * 256 + threadIdx.x; idx < total; idx += (long)gridDim.x * 256) {
        int c = (int)(idx % C);
        long t2 = idx / C;
        int n = (int)(t2 % Nt);
        int t = (int)(t2 / Nt);
        if (to_nt) dst[((long)n * T + t) * C + c] = src[((long)t * Nt + n) * C + c];
        else       dst[((long)t * Nt + n) * C + c] = src[((long)n * T + t) * C + c];
    }
}

__global__ void k_copy_rows(const float* src, float* dst, int nseq, int rows, int C,
                            long s_seq, int s_row0, long d_seq, int d_row0) {
    long total = (long)nseq * rows * C;
    for (long idx = (long)blockIdx.x * 256 + threadIdx.x; idx < total; idx += (long)gridDim.x * 256) {
        int c = (int)(idx % C);
        long t = idx / C;
        int r = (int)(t % rows);
        int sq = (int)(t / rows);
        dst[(long)sq * d_seq + (long)(d_row0 + r) * C + c] =
            src[(long)sq * s_seq + (long)(s_row0 + r) * C + c];
    }
}

__global__ void k_local_gather(const float* real, float* patches, float* kb,
                               int BT, long r_seq, int row_off,
                               int dH, int dW, int nh, int nw, int C) {
    long total = (long)BT * nh * nw * 36 * C;
    for (long idx = (long)blockIdx.x * 256 + threadIdx.x; idx < total; idx += (long)gridDim.x * 256) {
        int c = (int)(idx % C);
        long t = idx / C;
        int tok = (int)(t % 36);
        long p = t / 36;
        int pw = (int)(p % nw);
        long p2 = p / nw;
        int ph = (int)(p2 % nh);
        int bt = (int)(p2 / nh);
        int i = tok / 6, j = tok % 6;
        int gh = ph * 6 + i, gw = pw * 6 + j;
        int valid = (gh < dH) && (gw < dW);
        float v = 0.0f;
        if (valid) v = real[(long)bt * r_seq + (long)(row_off + gh * dW + gw) * C + c];
        patches[idx] = v;
        if (c == 0) kb[t] = valid ? 0.0f : NEGBIG;
    }
}

__global__ void k_local_scatter(const float* patches, float* real,
                                int BT, long r_seq, int row_off,
                                int dH, int dW, int nh, int nw, int C) {
    long total = (long)BT * nh * nw * 36 * C;
    for (long idx = (long)blockIdx.x * 256 + threadIdx.x; idx < total; idx += (long)gridDim.x * 256) {
        int c = (int)(idx % C);
        long t = idx / C;
        int tok = (int)(t % 36);
        long p = t / 36;
        int pw = (int)(p % nw);
        long p2 = p / nw;
        int ph = (int)(p2 % nh);
        int bt = (int)(p2 / nh);
        int i = tok / 6, j = tok % 6;
        int gh = ph * 6 + i, gw = pw * 6 + j;
        if (gh < dH && gw < dW)
            real[(long)bt * r_seq + (long)(row_off + gh * dW + gw) * C + c] = patches[idx];
    }
}

// ---------------------------------------------------------------------------
// Host-side orchestration
// ---------------------------------------------------------------------------
struct BW {
    const float *b1, *b2, *bo;
    const u16 *w1T, *w2T, *wkvT, *woT, *wqT;
};

struct Scr {
    u16 *ln, *lnc, *qb, *kvb, *ob, *hid;
};

static void launch_gemm(hipStream_t st, const u16* A, long lda, const u16* B, long ldb,
                        int M, int N, int K, float alpha,
                        const float* bias, const float* res, long ldr,
                        float* Cf, long ldc, u16* Cb, long ldcb, int gelu) {
    GemmArgs g;
    g.A = A; g.B = B; g.bias = bias; g.res = res; g.Cf = Cf; g.Cb = Cb;
    g.lda = lda; g.ldb = ldb; g.ldr = ldr; g.ldc = ldc; g.ldcb = ldcb;
    g.M = M; g.N = N; g.K = K; g.gelu = gelu; g.alpha = alpha;
    dim3 gr((N + 127) / 128, (M + 127) / 128);
    k_gemm<<<gr, 256, 0, st>>>(g);
}

static void launch_attn(hipStream_t st, const u16* qb, long qseq, const u16* kvb, long kvseq,
                        u16* ob, long oseq, const float* kb, long kbseq,
                        int nseq, int Nq, int Nk) {
    AttnArgs a;
    a.Q = qb; a.KV = kvb; a.O = ob; a.kb = kb;
    a.q_seq = qseq; a.kv_seq = kvseq; a.o_seq = oseq; a.kb_seq = kbseq;
    a.q_row = 384; a.kv_row = 768; a.o_row = 384;
    a.k_off = 0; a.v_off = 384;
    a.Nq = Nq; a.Nk = Nk; a.hd = 48;
    a.scale = 0.14433756f;  // 1/sqrt(48)
    dim3 gr((Nq + 15) / 16, 8, nseq);
    k_attn<<<gr, 32, 0, st>>>(a);
}

static void attn_mlp_block(hipStream_t st, const BW& w, float* x, int nseq, int L,
                           const float* kb, const Scr& s) {
    int M = nseq * L;
    k_layernorm<<<M, 128, 0, st>>>(x, s.ln, 384);
    launch_gemm(st, s.ln, 384, w.wqT, 384, M, 384, 384, 1.0f, nullptr, nullptr, 0, nullptr, 0, s.qb, 384, 0);
    launch_gemm(st, s.ln, 384, w.wkvT, 384, M, 768, 384, 1.0f, nullptr, nullptr, 0, nullptr, 0, s.kvb, 768, 0);
    launch_attn(st, s.qb, (long)L * 384, s.kvb, (long)L * 768, s.ob, (long)L * 384, kb, L, nseq, L, L);
    launch_gemm(st, s.ob, 384, w.woT, 384, M, 384, 384, 1.0f, w.bo, x, 384, x, 384, nullptr, 0, 0);
    k_layernorm<<<M, 128, 0, st>>>(x, s.ln, 384);
    launch_gemm(st, s.ln, 384, w.w1T, 384, M, 1536, 384, 1.0f, w.b1, nullptr, 0, nullptr, 0, s.hid, 1536, 1);
    launch_gemm(st, s.hid, 1536, w.w2T, 1536, M, 384, 1536, 1.0f, w.b2, x, 384, x, 384, nullptr, 0, 0);
}

static void cross_mlp_block(hipStream_t st, const BW& w, float* x, int nseq, int Lq,
                            const float* ctx, int Lk, const float* kb, const Scr& s) {
    int M = nseq * Lq, Mc = nseq * Lk;
    k_layernorm<<<M, 128, 0, st>>>(x, s.ln, 384);
    k_layernorm<<<Mc, 128, 0, st>>>(ctx, s.lnc, 384);
    launch_gemm(st, s.ln, 384, w.wqT, 384, M, 384, 384, 1.0f, nullptr, nullptr, 0, nullptr, 0, s.qb, 384, 0);
    launch_gemm(st, s.lnc, 384, w.wkvT, 384, Mc, 768, 384, 1.0f, nullptr, nullptr, 0, nullptr, 0, s.kvb, 768, 0);
    launch_attn(st, s.qb, (long)Lq * 384, s.kvb, (long)Lk * 768, s.ob, (long)Lq * 384, kb, Lk, nseq, Lq, Lk);
    launch_gemm(st, s.ob, 384, w.woT, 384, M, 384, 384, 1.0f, w.bo, x, 384, x, 384, nullptr, 0, 0);
    k_layernorm<<<M, 128, 0, st>>>(x, s.ln, 384);
    launch_gemm(st, s.ln, 384, w.w1T, 384, M, 1536, 384, 1.0f, w.b1, nullptr, 0, nullptr, 0, s.hid, 1536, 1);
    launch_gemm(st, s.hid, 1536, w.w2T, 1536, M, 384, 1536, 1.0f, w.b2, x, 384, x, 384, nullptr, 0, 0);
}

extern "C" void kernel_launch(void* const* d_in, const int* in_sizes, int n_in,
                              void* d_out, int out_size, void* d_ws, size_t ws_size,
                              hipStream_t stream) {
    (void)in_sizes; (void)n_in; (void)out_size; (void)ws_size;

    // ---- model constants (from setup_inputs; scalar inputs hard-coded) ----
    const int T_ = 8, Nt = 1920, NS = 256;
    const int DHh = 40, DWw = 40, NH = 7, NW = 7;
    const int NPAT = 8 * NH * NW;        // 392
    const long M_NT  = (long)Nt * T_;    // 15360
    const long M_RE  = (long)T_ * 1856;  // 14848
    const long M_PT  = (long)NPAT * 36;  // 14112

    // ---- parameter indexing (jax pytree: dict keys sorted alphabetically) ----
    // top: input_tensor, attn_mask, [params leaves...], n_sparse, dH, dW
    // params sorted: b_in, b_out, loc[6x8], p2v, sv, time, v2p, virtual, w_in, w_out
    // block sorted: b1, b2, bo, w1, w2, wkv, wo, wq
    const float* input   = (const float*)d_in[0];
    const unsigned char* mask = (const unsigned char*)d_in[1];
    const float* b_in    = (const float*)d_in[2];
    const float* b_out   = (const float*)d_in[3];
    const float* virtual_ = (const float*)d_in[244];
    const float* w_in_f  = (const float*)d_in[245];
    const float* w_out_f = (const float*)d_in[246];
    enum { LOC = 0, P2V = 1, SV = 2, TIME = 3, V2P = 4 };
    auto leaf = [&](int ty, int lv, int li) -> const float* {
        return (const float*)d_in[4 + ty * 48 + lv * 8 + li];
    };

    // ---- workspace bump allocator ----
    size_t off = 0;
    auto alloc = [&](size_t elems, size_t esz) -> void* {
        void* p = (char*)d_ws + off;
        off += (elems * esz + 255) & ~(size_t)255;
        return p;
    };

    const long BLK_ELEMS = 1769472;  // wq+wkv+wo+w1+w2 transposed bf16 per block
    u16* wbf    = (u16*)alloc(30L * BLK_ELEMS + 122880 + 49920, 2);
    u16* inbf   = (u16*)alloc(M_RE * 320, 2);
    float* tnt  = (float*)alloc(M_NT * 384, 4);
    float* tbt  = (float*)alloc(M_NT * 384, 4);
    float* realx = (float*)alloc(M_RE * 384, 4);
    float* virtx = (float*)alloc(512L * 384, 4);
    float* patch = (float*)alloc(M_PT * 384, 4);
    float* kbm   = (float*)alloc(M_RE, 4);
    float* kbl   = (float*)alloc(M_PT, 4);
    Scr s;
    s.ln  = (u16*)alloc(M_NT * 384, 2);
    s.lnc = (u16*)alloc(M_RE * 384, 2);
    s.qb  = (u16*)alloc(M_NT * 384, 2);
    s.kvb = (u16*)alloc(M_NT * 768, 2);
    s.ob  = (u16*)alloc(M_NT * 384, 2);
    s.hid = (u16*)alloc(M_NT * 1536, 2);

    static const long woffs[5] = {0, 147456, 442368, 589824, 1179648}; // wqT,wkvT,woT,w1T,w2T
    auto wptr = [&](int ty, int lv, int which) -> u16* {
        return wbf + (long)(ty * 6 + lv) * BLK_ELEMS + woffs[which];
    };
    auto transp = [&](const float* src, u16* dst, int R, int C) {
        dim3 gr((C + 31) / 32, (R + 31) / 32);
        k_transp_bf16<<<gr, 256, 0, stream>>>(src, dst, R, C);
    };

    // ---- weights -> bf16, transposed (B stored as N x K) ----
    for (int ty = 0; ty < 5; ++ty) {
        for (int lv = 0; lv < 6; ++lv) {
            transp(leaf(ty, lv, 7), wptr(ty, lv, 0), 384, 384);    // wq
            transp(leaf(ty, lv, 5), wptr(ty, lv, 1), 384, 768);    // wkv
            transp(leaf(ty, lv, 6), wptr(ty, lv, 2), 384, 384);    // wo
            transp(leaf(ty, lv, 3), wptr(ty, lv, 3), 384, 1536);   // w1
            transp(leaf(ty, lv, 4), wptr(ty, lv, 4), 1536, 384);   // w2
        }
    }
    u16* w_inT  = wbf + 30L * BLK_ELEMS;
    u16* w_outT = w_inT + 122880;
    transp(w_in_f, w_inT, 320, 384);
    transp(w_out_f, w_outT, 384, 130);

    auto mkbw = [&](int ty, int lv) -> BW {
        BW w;
        w.b1 = leaf(ty, lv, 0); w.b2 = leaf(ty, lv, 1); w.bo = leaf(ty, lv, 2);
        w.w1T = wptr(ty, lv, 3); w.w2T = wptr(ty, lv, 4); w.wkvT = wptr(ty, lv, 1);
        w.woT = wptr(ty, lv, 2); w.wqT = wptr(ty, lv, 0);
        return w;
    };
    auto cpr = [&](const float* srcp, float* dstp, int ns, int rows,
                   long ss, int sr0, long ds, int dr0) {
        long tot = (long)ns * rows * 384;
        k_copy_rows<<<(int)((tot + 255) / 256), 256, 0, stream>>>(
            srcp, dstp, ns, rows, 384, ss, sr0, ds, dr0);
    };

    // ---- key-mask bias from attn_mask ----
    k_mask_bias<<<(int)((M_RE + 255) / 256), 256, 0, stream>>>(mask, kbm, M_RE);

    // ---- input projection: real = input @ w_in + b_in ----
    k_cvt_bf16<<<(int)((M_RE * 320 + 255) / 256), 256, 0, stream>>>(input, inbf, M_RE * 320);
    launch_gemm(stream, inbf, 320, w_inT, 320, (int)M_RE, 384, 320, 1.0f,
                b_in, nullptr, 0, realx, 384, nullptr, 0, 0);

    // virtual tokens broadcast to each time step
    cpr(virtual_, virtx, 8, 64, 0, 0, 64L * 384, 0);

    // assemble [T][Nt][H] then permute to [Nt][T][H]
    cpr(realx, tbt, 8, 1856, 1856L * 384, 0, 1920L * 384, 0);
    cpr(virtx, tbt, 8, 64, 64L * 384, 0, 1920L * 384, 1856);
    {
        long tot = M_NT * 384;
        k_permute<<<(int)((tot + 255) / 256), 256, 0, stream>>>(tbt, tnt, T_, Nt, 384, 1);
    }

    // ---- 6 levels ----
    for (int lv = 0; lv < 6; ++lv) {
        // time attention over T=8 per (token)
        attn_mlp_block(stream, mkbw(TIME, lv), tnt, 1920, 8, nullptr, s);

        long tot = M_NT * 384;
        k_permute<<<(int)((tot + 255) / 256), 256, 0, stream>>>(tnt, tbt, T_, Nt, 384, 0);
        cpr(tbt, realx, 8, 1856, 1920L * 384, 0, 1856L * 384, 0);
        cpr(tbt, virtx, 8, 64, 1920L * 384, 1856, 64L * 384, 0);

        // virt <- cross(virt, real) with key mask, then self-attn on virt
        cross_mlp_block(stream, mkbw(V2P, lv), virtx, 8, 64, realx, 1856, kbm, s);
        attn_mlp_block(stream, mkbw(SV, lv), virtx, 8, 64, nullptr, s);

        // local attention over 6x6 patches of the dense region
        {
            long gt = (long)8 * NH * NW * 36 * 384;
            k_local_gather<<<(int)((gt + 255) / 256), 256, 0, stream>>>(
                realx, patch, kbl, 8, 1856L * 384, NS, DHh, DWw, NH, NW, 384);
            attn_mlp_block(stream, mkbw(LOC, lv), patch, NPAT, 36, kbl, s);
            k_local_scatter<<<(int)((gt + 255) / 256), 256, 0, stream>>>(
                patch, realx, 8, 1856L * 384, NS, DHh, DWw, NH, NW, 384);
        }

        // real <- cross(real, virt)  (query-side mask is a softmax no-op)
        cross_mlp_block(stream, mkbw(P2V, lv), realx, 8, 1856, virtx, 64, nullptr, s);

        cpr(realx, tbt, 8, 1856, 1856L * 384, 0, 1920L * 384, 0);
        cpr(virtx, tbt, 8, 64, 64L * 384, 0, 1920L * 384, 1856);
        if (lv < 5)
            k_permute<<<(int)((tot + 255) / 256), 256, 0, stream>>>(tbt, tnt, T_, Nt, 384, 1);
    }

    // ---- output head: out = real @ w_out + b_out ----
    k_cvt_bf16<<<(int)((M_RE * 384 + 255) / 256), 256, 0, stream>>>(realx, s.ln, M_RE * 384);
    launch_gemm(stream, s.ln, 384, w_outT, 384, (int)M_RE, 130, 384, 1.0f,
                b_out, nullptr, 0, (float*)d_out, 130, nullptr, 0, 0);
}
